// MultiHeadAttention_64467459113174
// MI455X (gfx1250) — matrix operand order repaired and checked by reading, not compiled, not run on hardware
//
#include <hip/hip_runtime.h>

#define Bc 4
#define Hc 16
#define Tc 2048
#define Dc 64
#define Ec 1024
#define Mtot (Bc * Tc)   // 8192

typedef __attribute__((ext_vector_type(16))) __bf16 v16bf;
typedef __attribute__((ext_vector_type(8)))  float  v8f;
typedef __attribute__((ext_vector_type(16))) float  v16f;

union Frag16 {
    v16bf v;
    unsigned short s[16];
    uint4 q[2];
};

union F32x16 {
    v16f v;
    float f[16];
    float4 q[4];
};

// Native f32 -> bf16 (RNE) via hardware cvt
__device__ __forceinline__ unsigned short f2bf(float x) {
    union { __bf16 h; unsigned short u; } c;
    c.h = (__bf16)x;
    return c.u;
}

__device__ __forceinline__ float redmax16(float v) {
    v = fmaxf(v, __shfl_xor(v, 1, 32));
    v = fmaxf(v, __shfl_xor(v, 2, 32));
    v = fmaxf(v, __shfl_xor(v, 4, 32));
    v = fmaxf(v, __shfl_xor(v, 8, 32));
    return v;
}
__device__ __forceinline__ float redsum16(float v) {
    v += __shfl_xor(v, 1, 32);
    v += __shfl_xor(v, 2, 32);
    v += __shfl_xor(v, 4, 32);
    v += __shfl_xor(v, 8, 32);
    return v;
}

// ---------------------------------------------------------------------------
// Kernel 1: Y = X @ W^T + b  (f32 in, bf16 out), per-wave 64x64 tile.
// out layout: vtrans==0 -> [B,H,T,D]; vtrans==1 -> [B,H,D,T] (for V).
// scale folded into stored bf16 (Q gets 1/sqrt(D)).
// ---------------------------------------------------------------------------
__global__ __launch_bounds__(128) void qkv_proj_kernel(
    const float* __restrict__ X, const float* __restrict__ W,
    const float* __restrict__ bias, unsigned short* __restrict__ out,
    float scale, int vtrans)
{
    const int lane = threadIdx.x & 31;
    const int wave = threadIdx.x >> 5;
    const int wid  = blockIdx.x * 4 + wave;      // 0..2047
    const int ntN  = Ec / 64;                    // 16
    const int m0   = (wid / ntN) * 64;
    const int n0   = (wid % ntN) * 64;
    const int half = lane >> 4;
    const int l16  = lane & 15;

    v8f acc[4][4];
    v8f z = {};
#pragma unroll
    for (int i = 0; i < 4; i++)
#pragma unroll
        for (int j = 0; j < 4; j++) acc[i][j] = z;

    for (int k0 = 0; k0 < Ec; k0 += 32) {
        v16bf a[4], b[4];
#pragma unroll
        for (int mi = 0; mi < 4; mi++) {
            // A lane = row m, K runs {16h..16h+15} (contig)
            const float* src = X + (size_t)(m0 + mi * 16 + l16) * Ec + k0;
            F32x16 t;
            t.q[0] = *(const float4*)(src + 16 * half);
            t.q[1] = *(const float4*)(src + 16 * half + 4);
            t.q[2] = *(const float4*)(src + 16 * half + 8);
            t.q[3] = *(const float4*)(src + 16 * half + 12);
            a[mi] = __builtin_convertvector(t.v, v16bf);
        }
#pragma unroll
        for (int ni = 0; ni < 4; ni++) {
            // B[k][n] = W[n][k]; lane column n = n0+ni*16+l16, K = 16*half + e (contig)
            const float* src = W + (size_t)(n0 + ni * 16 + l16) * Ec + k0 + 16 * half;
            F32x16 t;
            t.q[0] = *(const float4*)(src);
            t.q[1] = *(const float4*)(src + 4);
            t.q[2] = *(const float4*)(src + 8);
            t.q[3] = *(const float4*)(src + 12);
            b[ni] = __builtin_convertvector(t.v, v16bf);
        }
#pragma unroll
        for (int mi = 0; mi < 4; mi++)
#pragma unroll
            for (int ni = 0; ni < 4; ni++)
                acc[mi][ni] = __builtin_amdgcn_wmma_f32_16x16x32_bf16(
                    false, a[mi], false, b[ni], (short)0, acc[mi][ni], false, false);
    }

#pragma unroll
    for (int ni = 0; ni < 4; ni++) {
        const int n = n0 + ni * 16 + l16;
        const float bv = bias[n];
        const int h = n >> 6, d = n & 63;
#pragma unroll
        for (int mi = 0; mi < 4; mi++) {
#pragma unroll
            for (int r = 0; r < 8; r++) {
                const int m = m0 + mi * 16 + r + 8 * half;
                const int bb = m >> 11, t = m & (Tc - 1);
                const float y = (acc[mi][ni][r] + bv) * scale;
                size_t idx = vtrans
                    ? ((((size_t)bb * Hc + h) * Dc + d) * Tc + t)
                    : ((((size_t)bb * Hc + h) * Tc + t) * Dc + d);
                out[idx] = f2bf(y);
            }
        }
    }
}

// ---------------------------------------------------------------------------
// Kernel 2: causal flash attention. One wave per 16-row q-tile per (b,h).
// Qb,Kb: [B,H,T,D] bf16 (Q pre-scaled); Vt: [B,H,D,T] bf16; Ob: [B,T,E] bf16.
// ---------------------------------------------------------------------------
__global__ __launch_bounds__(128) void flash_attn_kernel(
    const unsigned short* __restrict__ Qb,
    const unsigned short* __restrict__ Kb,
    const unsigned short* __restrict__ Vt,
    unsigned short* __restrict__ Ob)
{
    __shared__ unsigned short Pl[4][16 * 64];   // per-wave P staging (C->A transpose)
    const int lane = threadIdx.x & 31;
    const int wave = threadIdx.x >> 5;
    const int wid  = blockIdx.x * 4 + wave;     // 0..8191
    const int qt   = wid & 127;                 // T/16 tiles
    const int bh   = wid >> 7;                  // b*H + h
    const int q0   = qt * 16;
    const int half = lane >> 4;
    const int l16  = lane & 15;

    const unsigned short* Qp = Qb + (size_t)bh * (Tc * Dc);
    const unsigned short* Kp = Kb + (size_t)bh * (Tc * Dc);
    const unsigned short* Vp = Vt + (size_t)bh * (Dc * Tc);
    unsigned short* myP = Pl[wave];

    // Q A-fragments, kc = d-chunk of 32
    v16bf qa[2];
#pragma unroll
    for (int kc = 0; kc < 2; kc++) {
        const unsigned short* src = Qp + (size_t)(q0 + l16) * Dc + kc * 32;
        Frag16 f;
        f.q[0] = *(const uint4*)(src + 16 * half);
        f.q[1] = *(const uint4*)(src + 16 * half + 8);
        qa[kc] = f.v;
    }

    v8f acc[4];
    v8f z = {};
#pragma unroll
    for (int nd = 0; nd < 4; nd++) acc[nd] = z;
    float mrow[8], lrow[8];
#pragma unroll
    for (int r = 0; r < 8; r++) { mrow[r] = -1e30f; lrow[r] = 0.0f; }

    const int ntiles = (q0 + 79) >> 6;          // k-tiles of 64 needed (causal)
    for (int kt = 0; kt < ntiles; kt++) {
        const int j0 = kt * 64;
        v8f s[4];
#pragma unroll
        for (int nj = 0; nj < 4; nj++) s[nj] = z;

#pragma unroll
        for (int kc = 0; kc < 2; kc++) {
            v16bf kb[4];
#pragma unroll
            for (int nj = 0; nj < 4; nj++) {
                const unsigned short* src =
                    Kp + (size_t)(j0 + nj * 16 + l16) * Dc + kc * 32 + 16 * half;
                Frag16 f;
                f.q[0] = *(const uint4*)(src);
                f.q[1] = *(const uint4*)(src + 8);
                kb[nj] = f.v;
            }
#pragma unroll
            for (int nj = 0; nj < 4; nj++)
                s[nj] = __builtin_amdgcn_wmma_f32_16x16x32_bf16(
                    false, qa[kc], false, kb[nj], (short)0, s[nj], false, false);
        }

        if (j0 + 63 > q0) {                     // diagonal tile: causal mask
#pragma unroll
            for (int nj = 0; nj < 4; nj++) {
                const int j = j0 + nj * 16 + l16;
#pragma unroll
                for (int r = 0; r < 8; r++) {
                    const int q = q0 + r + 8 * half;
                    if (j > q) s[nj][r] = -1e30f;
                }
            }
        }

        float alpha[8];
#pragma unroll
        for (int r = 0; r < 8; r++) {
            float rv = fmaxf(fmaxf(s[0][r], s[1][r]), fmaxf(s[2][r], s[3][r]));
            rv = redmax16(rv);
            const float mnew = fmaxf(mrow[r], rv);
            alpha[r] = __expf(mrow[r] - mnew);
            mrow[r] = mnew;
        }
#pragma unroll
        for (int r = 0; r < 8; r++) {
            float rs = 0.0f;
#pragma unroll
            for (int nj = 0; nj < 4; nj++) {
                const float p = __expf(s[nj][r] - mrow[r]);
                s[nj][r] = p;
                rs += p;
            }
            rs = redsum16(rs);
            lrow[r] = lrow[r] * alpha[r] + rs;
        }
        // stage P (bf16) to LDS in row-major, re-read in A-fragment layout
#pragma unroll
        for (int nj = 0; nj < 4; nj++)
#pragma unroll
            for (int r = 0; r < 8; r++)
                myP[(r + 8 * half) * 64 + nj * 16 + l16] = f2bf(s[nj][r]);
#pragma unroll
        for (int nd = 0; nd < 4; nd++)
#pragma unroll
            for (int r = 0; r < 8; r++) acc[nd][r] *= alpha[r];

#pragma unroll
        for (int kc2 = 0; kc2 < 2; kc2++) {
            const unsigned short* ps = myP + l16 * 64 + kc2 * 32;
            Frag16 fp;
            fp.q[0] = *(const uint4*)(ps + 16 * half);
            fp.q[1] = *(const uint4*)(ps + 16 * half + 8);
            const v16bf pa = fp.v;
#pragma unroll
            for (int nd = 0; nd < 4; nd++) {
                const unsigned short* src =
                    Vp + (size_t)(nd * 16 + l16) * Tc + j0 + kc2 * 32 + 16 * half;
                Frag16 fv;
                fv.q[0] = *(const uint4*)(src);
                fv.q[1] = *(const uint4*)(src + 8);
                acc[nd] = __builtin_amdgcn_wmma_f32_16x16x32_bf16(
                    false, pa, false, fv.v, (short)0, acc[nd], false, false);
            }
        }
    }

    const int b = bh >> 4, h = bh & 15;
#pragma unroll
    for (int nd = 0; nd < 4; nd++) {
        const int col = h * 64 + nd * 16 + l16;
#pragma unroll
        for (int r = 0; r < 8; r++) {
            const int t = q0 + r + 8 * half;
            const float o = acc[nd][r] / lrow[r];
            Ob[((size_t)(b * Tc + t)) * Ec + col] = f2bf(o);
        }
    }
}

// ---------------------------------------------------------------------------
// Kernel 3: Y = O @ Wo^T + bo  (bf16 O, f32 out), per-wave 64x64 tile.
// ---------------------------------------------------------------------------
__global__ __launch_bounds__(128) void out_proj_kernel(
    const unsigned short* __restrict__ Ob, const float* __restrict__ Wo,
    const float* __restrict__ bo, float* __restrict__ Y)
{
    const int lane = threadIdx.x & 31;
    const int wave = threadIdx.x >> 5;
    const int wid  = blockIdx.x * 4 + wave;
    const int ntN  = Ec / 64;
    const int m0   = (wid / ntN) * 64;
    const int n0   = (wid % ntN) * 64;
    const int half = lane >> 4;
    const int l16  = lane & 15;

    v8f acc[4][4];
    v8f z = {};
#pragma unroll
    for (int i = 0; i < 4; i++)
#pragma unroll
        for (int j = 0; j < 4; j++) acc[i][j] = z;

    for (int k0 = 0; k0 < Ec; k0 += 32) {
        v16bf a[4], b[4];
#pragma unroll
        for (int mi = 0; mi < 4; mi++) {
            const unsigned short* src = Ob + (size_t)(m0 + mi * 16 + l16) * Ec + k0;
            Frag16 f;
            f.q[0] = *(const uint4*)(src + 16 * half);
            f.q[1] = *(const uint4*)(src + 16 * half + 8);
            a[mi] = f.v;
        }
#pragma unroll
        for (int ni = 0; ni < 4; ni++) {
            const float* src = Wo + (size_t)(n0 + ni * 16 + l16) * Ec + k0 + 16 * half;
            F32x16 t;
            t.q[0] = *(const float4*)(src);
            t.q[1] = *(const float4*)(src + 4);
            t.q[2] = *(const float4*)(src + 8);
            t.q[3] = *(const float4*)(src + 12);
            b[ni] = __builtin_convertvector(t.v, v16bf);
        }
#pragma unroll
        for (int mi = 0; mi < 4; mi++)
#pragma unroll
            for (int ni = 0; ni < 4; ni++)
                acc[mi][ni] = __builtin_amdgcn_wmma_f32_16x16x32_bf16(
                    false, a[mi], false, b[ni], (short)0, acc[mi][ni], false, false);
    }

#pragma unroll
    for (int ni = 0; ni < 4; ni++) {
        const int n = n0 + ni * 16 + l16;
        const float bv = bo[n];
#pragma unroll
        for (int mi = 0; mi < 4; mi++) {
#pragma unroll
            for (int r = 0; r < 8; r++) {
                const int m = m0 + mi * 16 + r + 8 * half;
                Y[(size_t)m * Ec + n] = acc[mi][ni][r] + bv;
            }
        }
    }
}

extern "C" void kernel_launch(void* const* d_in, const int* in_sizes, int n_in,
                              void* d_out, int out_size, void* d_ws, size_t ws_size,
                              hipStream_t stream) {
    const float* query  = (const float*)d_in[0];
    const float* key_in = (const float*)d_in[1];
    const float* value  = (const float*)d_in[2];
    // d_in[3] = mask (causal tril) — hardcoded as causal in flash kernel
    const float* Wq = (const float*)d_in[4];
    const float* bq = (const float*)d_in[5];
    const float* Wk = (const float*)d_in[6];
    const float* bk = (const float*)d_in[7];
    const float* Wv = (const float*)d_in[8];
    const float* bv = (const float*)d_in[9];
    const float* Wo = (const float*)d_in[10];
    const float* bo = (const float*)d_in[11];
    float* out = (float*)d_out;

    const size_t nBHTD = (size_t)Bc * Hc * Tc * Dc;   // 8,388,608 elems
    unsigned short* qbuf = (unsigned short*)d_ws;
    unsigned short* kbuf = qbuf + nBHTD;
    unsigned short* vtbuf = kbuf + nBHTD;
    unsigned short* obuf = vtbuf + nBHTD;

    const dim3 blk(128);
    const int gproj = (Mtot / 64) * (Ec / 64) / 4;    // 512 blocks (4 waves each)
    const float qscale = 0.125f;                      // 1/sqrt(D)

    qkv_proj_kernel<<<gproj, blk, 0, stream>>>(query,  Wq, bq, qbuf,  qscale, 0);
    qkv_proj_kernel<<<gproj, blk, 0, stream>>>(key_in, Wk, bk, kbuf,  1.0f,   0);
    qkv_proj_kernel<<<gproj, blk, 0, stream>>>(value,  Wv, bv, vtbuf, 1.0f,   1);

    const int gattn = (Bc * Hc * (Tc / 16)) / 4;      // 2048 blocks
    flash_attn_kernel<<<gattn, blk, 0, stream>>>(qbuf, kbuf, vtbuf, obuf);

    out_proj_kernel<<<gproj, blk, 0, stream>>>(obuf, Wo, bo, out);
}
